// SE_Block_50611894616457
// MI455X (gfx1250) — matrix operand order repaired and checked by reading, not compiled, not run on hardware
//
#include <hip/hip_runtime.h>
#include <hip/hip_fp16.h>
#include <math.h>

// SE block: B=32, H=W=56, C=256, R=16.
// Memory-bound problem (~205 MB min HBM traffic -> ~9us at 23.3 TB/s).
// WMMA (v_wmma_f32_16x16x32_f16) used for the 256->16->256 excitation MLP.

typedef __attribute__((ext_vector_type(16))) _Float16 v16h;
typedef __attribute__((ext_vector_type(8)))  float    v8f;

#define B_  32
#define H_  56
#define W_  56
#define C_  256
#define CR_ 16          // C / ratio
#define HW_ (H_ * W_)   // 3136

// ---------------------------------------------------------------------------
// Phase 1a: per-(b,h) partial sums over W. grid = B*H blocks, 256 thr (1/chan)
// Coalesced: 256 consecutive threads read 256 consecutive channels.
// ---------------------------------------------------------------------------
__global__ void se_rowsum(const float* __restrict__ in, float* __restrict__ partial)
{
    const int bh = blockIdx.x;          // b*H_ + h
    const int c  = threadIdx.x;
    const float* p = in + (size_t)bh * (W_ * C_) + c;
    float acc = 0.0f;
#pragma unroll 8
    for (int w = 0; w < W_; ++w) acc += p[w * C_];
    partial[(size_t)bh * C_ + c] = acc;
}

// ---------------------------------------------------------------------------
// Phase 1b: reduce the H partials -> squeeze sums [B,C]. grid = B, 256 thr.
// Deterministic (no float atomics).
// ---------------------------------------------------------------------------
__global__ void se_colreduce(const float* __restrict__ partial, float* __restrict__ squeeze)
{
    const int b = blockIdx.x, c = threadIdx.x;
    const float* p = partial + (size_t)b * H_ * C_ + c;
    float acc = 0.0f;
#pragma unroll 8
    for (int h = 0; h < H_; ++h) acc += p[h * C_];
    squeeze[b * C_ + c] = acc;          // raw sum; divided by HW in the MLP
}

// ---------------------------------------------------------------------------
// Phase 2: excitation MLP on one wave32.
//   h = relu(mean(S) @ W1 + b1)   [32,16]   -> 2 M-tiles x 8 K-steps WMMA
//   E = sigmoid(h @ W2 + b2)      [32,256]  -> 2 M-tiles x 16 N-tiles WMMA
// Fragment layouts used here (wave32):
//   A(16x32,f16): lane row m = lane&15; elem e -> k = e + 16*(lane>=16)
//   B(32x16,f16): lane col n = lane&15; elem e -> k = e + 16*(lane>=16) (same as A; not ISA table order)
//   C/D(16x16,f32): lane col n = lane&15; elem r -> m = r + 8*(lane>=16)
// ---------------------------------------------------------------------------
__global__ void __launch_bounds__(32)
se_mlp(const float* __restrict__ squeeze,   // [32,256] sums
       const float* __restrict__ W1,        // [256,16]
       const float* __restrict__ b1,        // [16]
       const float* __restrict__ W2,        // [16,256]
       const float* __restrict__ b2,        // [256]
       float* __restrict__ excite)          // [32,256]
{
    const int lane = threadIdx.x;           // 0..31, EXEC all ones (WMMA req.)
    const int half = lane >> 4;             // 0 or 1
    const int lmod = lane & 15;
    const float inv_hw = 1.0f / (float)HW_;

    __shared__ _Float16 hbuf[32][16];       // relu(h), re-read in A layout

    // ---------------- GEMM1: [32,256] x [256,16] ----------------
    for (int mt = 0; mt < 2; ++mt) {
        v8f acc;
        const float bias1 = b1[lmod];       // bias depends on n = lmod only
#pragma unroll
        for (int r = 0; r < 8; ++r) acc[r] = bias1;

        const int m = mt * 16 + lmod;       // A row owned by this lane
        for (int kk = 0; kk < 8; ++kk) {    // K = 256 in steps of 32
            const int kbase = kk * 32;
            v16h a, b;
#pragma unroll
            for (int e = 0; e < 16; ++e) {
                const int k = kbase + e + (half ? 16 : 0);
                a[e] = (_Float16)(squeeze[m * C_ + k] * inv_hw);
            }
#pragma unroll
            for (int e = 0; e < 16; ++e) {
                const int k = kbase + e + (half ? 16 : 0);
                b[e] = (_Float16)W1[k * CR_ + lmod];
            }
            acc = __builtin_amdgcn_wmma_f32_16x16x32_f16(
                false, a, false, b, (short)0, acc, false, false);
        }
        // relu -> LDS; D elem r is (m = mt*16 + r + 8*half, n = lmod)
#pragma unroll
        for (int r = 0; r < 8; ++r) {
            float v = acc[r];
            hbuf[mt * 16 + r + 8 * half][lmod] = (_Float16)(v > 0.0f ? v : 0.0f);
        }
    }
    __syncthreads();

    // ---------------- GEMM2: [32,16] x [16,256], K padded 16->32 ----------------
    for (int mt = 0; mt < 2; ++mt) {
        v16h a;
        const int m = mt * 16 + lmod;
#pragma unroll
        for (int e = 0; e < 16; ++e) {
            const int k = e + (half ? 16 : 0);   // 0..31
            a[e] = (k < 16) ? hbuf[m][k] : (_Float16)0.0f;
        }
        for (int nt = 0; nt < 16; ++nt) {
            const int n = nt * 16 + lmod;
            v16h b;
#pragma unroll
            for (int e = 0; e < 16; ++e) {
                const int k = e + (half ? 16 : 0);
                b[e] = (k < 16) ? (_Float16)W2[k * C_ + n] : (_Float16)0.0f;
            }
            v8f acc;
            const float bias2 = b2[n];
#pragma unroll
            for (int r = 0; r < 8; ++r) acc[r] = bias2;

            acc = __builtin_amdgcn_wmma_f32_16x16x32_f16(
                false, a, false, b, (short)0, acc, false, false);

#pragma unroll
            for (int r = 0; r < 8; ++r) {
                const int mm = mt * 16 + r + 8 * half;
                excite[mm * C_ + n] = 1.0f / (1.0f + __expf(-acc[r]));
            }
        }
    }
}

// ---------------------------------------------------------------------------
// Phase 3: out = in * E[b,c], float4-vectorized (c%4==0 within a float4 since
// C=256). Input should still be L2-resident (102.8 MB < 192 MB L2).
// ---------------------------------------------------------------------------
__global__ void se_scale(const float4* __restrict__ in4,
                         const float*  __restrict__ excite,
                         float4* __restrict__ out4, unsigned total4)
{
    const unsigned i = blockIdx.x * blockDim.x + threadIdx.x;
    if (i >= total4) return;
    const unsigned flat = i * 4u;                    // element index
    const unsigned c    = flat & (C_ - 1);           // channel base (mult of 4)
    const unsigned b    = flat / (unsigned)(HW_ * C_);
    const float4 v = in4[i];
    const float4 e = *(const float4*)(excite + b * C_ + c);
    float4 o;
    o.x = v.x * e.x; o.y = v.y * e.y; o.z = v.z * e.z; o.w = v.w * e.w;
    out4[i] = o;
}

// ---------------------------------------------------------------------------
extern "C" void kernel_launch(void* const* d_in, const int* in_sizes, int n_in,
                              void* d_out, int out_size, void* d_ws, size_t ws_size,
                              hipStream_t stream)
{
    const float* in = (const float*)d_in[0];   // [32,56,56,256]
    const float* W1 = (const float*)d_in[1];   // [256,16]
    const float* b1 = (const float*)d_in[2];   // [16]
    const float* W2 = (const float*)d_in[3];   // [16,256]
    const float* b2 = (const float*)d_in[4];   // [256]
    float* out = (float*)d_out;

    // Workspace layout (floats): partial[B*H*C] | squeeze[B*C] | excite[B*C]
    float* partial = (float*)d_ws;                     // 458752 floats
    float* squeeze = partial + (size_t)B_ * H_ * C_;   // 8192 floats
    float* excite  = squeeze + (size_t)B_ * C_;        // 8192 floats
    (void)in_sizes; (void)n_in; (void)out_size; (void)ws_size;

    se_rowsum  <<<B_ * H_, C_, 0, stream>>>(in, partial);
    se_colreduce<<<B_,     C_, 0, stream>>>(partial, squeeze);
    se_mlp     <<<1,       32, 0, stream>>>(squeeze, W1, b1, W2, b2, excite);

    const unsigned total4 = (unsigned)((size_t)B_ * HW_ * C_ / 4);
    se_scale   <<<(total4 + 255) / 256, 256, 0, stream>>>(
        (const float4*)in, excite, (float4*)out, total4);
}
